// MultiHeadSelfAttention_41618233098685
// MI455X (gfx1250) — matrix operand order repaired and checked by reading, not compiled, not run on hardware
//
#include <hip/hip_runtime.h>

typedef __attribute__((ext_vector_type(16))) __bf16 bfvec16;
typedef __attribute__((ext_vector_type(8)))  float  v8f;
using ushort = unsigned short;

// ---------------- problem constants ----------------
constexpr int DIM   = 768;
constexpr int NH    = 12;
constexpr int HD    = 64;
constexpr int NSEQ  = 2048;
constexpr int BATCH = 2;
constexpr int ROWS  = BATCH * NSEQ;   // 4096
constexpr int QKVN  = 3 * DIM;        // 2304
constexpr int BH    = BATCH * NH;     // 24

// ---------------- helpers ----------------
__device__ __forceinline__ ushort f2bf(float f) {
    unsigned u = __float_as_uint(f);
    unsigned r = u + 0x7FFFu + ((u >> 16) & 1u);   // round-to-nearest-even
    return (ushort)(r >> 16);
}

// A-fragment (16x32, 16-bit): lane&15 = row M, K chunks at 8*grp and 16+8*grp
__device__ __forceinline__ bfvec16 load_a_frag(const ushort* base, int stride) {
    int lane = threadIdx.x & 31;
    const ushort* p = base + (lane & 15) * stride + ((lane >> 4) << 3);
    bfvec16 f;
    uint4* fp = reinterpret_cast<uint4*>(&f);
    fp[0] = *reinterpret_cast<const uint4*>(p);
    fp[1] = *reinterpret_cast<const uint4*>(p + 16);
    return f;
}

// B-fragment (32x16, 16-bit), source stored as N x K row-major:
// lane&15 = column N, K chunks at 8*grp and 16+8*grp
__device__ __forceinline__ bfvec16 load_b_frag(const ushort* base, int stride) {
    int lane = threadIdx.x & 31;
    const ushort* p = base + (lane & 15) * stride + ((lane >> 4) << 3);
    bfvec16 f;
    uint4* fp = reinterpret_cast<uint4*>(&f);
    fp[0] = *reinterpret_cast<const uint4*>(p);
    fp[1] = *reinterpret_cast<const uint4*>(p + 16);
    return f;
}

__device__ __forceinline__ v8f wmma_bf16(bfvec16 a, bfvec16 b, v8f c) {
    return __builtin_amdgcn_wmma_f32_16x16x32_bf16(
        false, a, false, b, (short)0, c, false, false);
}

// ---------------- conversion kernels ----------------
__global__ void __launch_bounds__(256) cvt_bf16_kernel(const float* __restrict__ src,
                                                       ushort* __restrict__ dst, int n) {
    int i = blockIdx.x * 256 + threadIdx.x;
    if (i < n) dst[i] = f2bf(src[i]);
}

// src is K x N (row-major); dst is N x K (row-major)  -> B-operand layout
__global__ void __launch_bounds__(256) cvt_transpose_kernel(const float* __restrict__ src,
                                                            ushort* __restrict__ dst,
                                                            int K, int N) {
    int i = blockIdx.x * 256 + threadIdx.x;
    if (i < K * N) {
        int n = i / K;
        int k = i - n * K;
        dst[n * K + k] = f2bf(src[k * N + n]);
    }
}

// ---------------- QKV projection GEMM ----------------
// 64x64 output tile per wave: 16 WMMAs per k=32 step against 8 fragment loads.
// xbf[ROWS x DIM] @ w_qkvT[QKVN x DIM]^T + b -> Q,K [B,H,N,Hd] (Q pre-scaled by 1/8),
// V transposed [B,H,Hd,N]
__global__ void __launch_bounds__(256) gemm_qkv_kernel(const ushort* __restrict__ xbf,
                                                       const ushort* __restrict__ wT,
                                                       const float*  __restrict__ bias,
                                                       ushort* __restrict__ Qb,
                                                       ushort* __restrict__ Kb,
                                                       ushort* __restrict__ Vt) {
    constexpr int TILES_N = QKVN / 64;  // 36
    int wave = (blockIdx.x * blockDim.x + threadIdx.x) >> 5;
    int lane = threadIdx.x & 31;
    int tm = wave / TILES_N, tn = wave - tm * TILES_N;
    int m0 = tm * 64, n0 = tn * 64;

    v8f acc[4][4] = {};
    for (int kk = 0; kk < DIM; kk += 32) {
        bfvec16 a[4], b[4];
#pragma unroll
        for (int i = 0; i < 4; ++i)
            a[i] = load_a_frag(xbf + (m0 + 16 * i) * DIM + kk, DIM);
#pragma unroll
        for (int j = 0; j < 4; ++j)
            b[j] = load_b_frag(wT + (n0 + 16 * j) * DIM + kk, DIM);
#pragma unroll
        for (int i = 0; i < 4; ++i)
#pragma unroll
            for (int j = 0; j < 4; ++j)
                acc[i][j] = wmma_bf16(a[i], b[j], acc[i][j]);
    }

    int cn   = lane & 15;
    int rgrp = (lane >> 4) * 8;
#pragma unroll
    for (int i = 0; i < 4; ++i)
#pragma unroll
        for (int j = 0; j < 4; ++j) {
            int c   = n0 + j * 16 + cn;
            float bv = bias[c];
            int t   = c / DIM;        // uniform per tile (64 | 768)
            int rem = c - t * DIM;
            int h   = rem >> 6;
            int hd  = rem & 63;
#pragma unroll
            for (int r = 0; r < 8; ++r) {
                int row = m0 + i * 16 + rgrp + r;
                int b2  = row >> 11;       // row / NSEQ
                int n   = row & (NSEQ - 1);
                float val = acc[i][j][r] + bv;
                if (t == 0) {
                    Qb[(((b2 * NH + h) * NSEQ) + n) * HD + hd] = f2bf(val * 0.125f);
                } else if (t == 1) {
                    Kb[(((b2 * NH + h) * NSEQ) + n) * HD + hd] = f2bf(val);
                } else {
                    Vt[(((b2 * NH + h) * HD) + hd) * NSEQ + n] = f2bf(val);
                }
            }
        }
}

// ---------------- flash attention ----------------
// One wave per 32 query rows (two 16-row tiles): per 32-key step,
// 8 S-WMMAs + 8 PV-WMMAs against 4 K-fragments + 4 V-fragments.
// grid.x = BH * (NSEQ/256), 8 waves/block.
__global__ void __launch_bounds__(256) attn_kernel(const ushort* __restrict__ Qb,
                                                   const ushort* __restrict__ Kb,
                                                   const ushort* __restrict__ Vt,
                                                   ushort* __restrict__ Ob) {
    __shared__ ushort ldsP[8][32 * 32];   // 2 KB per wave

    constexpr int QBLK = NSEQ / 256;  // 8 q-blocks of 256 rows per (b,h)
    int waveInBlk = threadIdx.x >> 5;
    int lane      = threadIdx.x & 31;
    int bh = blockIdx.x / QBLK;
    int q0 = (blockIdx.x - bh * QBLK) * 256 + waveInBlk * 32;

    const ushort* qptr = Qb + (bh * NSEQ + q0) * HD;
    const ushort* kptr = Kb + bh * NSEQ * HD;
    const ushort* vptr = Vt + bh * HD * NSEQ;

    // Q fragments: [row-tile][channel chunk]
    bfvec16 aq[2][2];
#pragma unroll
    for (int rt = 0; rt < 2; ++rt)
#pragma unroll
        for (int cj = 0; cj < 2; ++cj)
            aq[rt][cj] = load_a_frag(qptr + rt * 16 * HD + cj * 32, HD);

    float mrun[2][8], lrun[2][8];
#pragma unroll
    for (int rt = 0; rt < 2; ++rt)
#pragma unroll
        for (int r = 0; r < 8; ++r) { mrun[rt][r] = -3.0e38f; lrun[rt][r] = 0.f; }
    v8f accO[2][4] = {};

    int cn   = lane & 15;
    int rgrp = (lane >> 4) * 8;
    ushort* myP = ldsP[waveInBlk];

    for (int kb = 0; kb < NSEQ; kb += 32) {
        // K fragments: [key-tile][channel chunk]
        bfvec16 bk[2][2];
#pragma unroll
        for (int kt = 0; kt < 2; ++kt)
#pragma unroll
            for (int cj = 0; cj < 2; ++cj)
                bk[kt][cj] = load_b_frag(kptr + (kb + 16 * kt) * HD + 32 * cj, HD);

        // S = Q @ K^T
        v8f s[2][2] = {};
#pragma unroll
        for (int rt = 0; rt < 2; ++rt)
#pragma unroll
            for (int kt = 0; kt < 2; ++kt) {
                s[rt][kt] = wmma_bf16(aq[rt][0], bk[kt][0], s[rt][kt]);
                s[rt][kt] = wmma_bf16(aq[rt][1], bk[kt][1], s[rt][kt]);
            }

        // online softmax per row-tile (reduced across 16-lane halves)
#pragma unroll
        for (int rt = 0; rt < 2; ++rt) {
#pragma unroll
            for (int r = 0; r < 8; ++r) {
                float t = fmaxf(s[rt][0][r], s[rt][1][r]);
                t = fmaxf(t, __shfl_xor(t, 1, 32));
                t = fmaxf(t, __shfl_xor(t, 2, 32));
                t = fmaxf(t, __shfl_xor(t, 4, 32));
                t = fmaxf(t, __shfl_xor(t, 8, 32));
                float mnew = fmaxf(mrun[rt][r], t);
                float al   = __expf(mrun[rt][r] - mnew);
                mrun[rt][r] = mnew;
                float p0 = __expf(s[rt][0][r] - mnew);
                float p1 = __expf(s[rt][1][r] - mnew);
                float rs = p0 + p1;
                rs += __shfl_xor(rs, 1, 32);
                rs += __shfl_xor(rs, 2, 32);
                rs += __shfl_xor(rs, 4, 32);
                rs += __shfl_xor(rs, 8, 32);
                lrun[rt][r] = lrun[rt][r] * al + rs;
#pragma unroll
                for (int c = 0; c < 4; ++c) accO[rt][c][r] *= al;
                int mrow = rt * 16 + rgrp + r;
                myP[mrow * 32 + cn]      = f2bf(p0);
                myP[mrow * 32 + 16 + cn] = f2bf(p1);
            }
        }
        __syncthreads();

        // O += P @ V  (P as A-fragments from LDS; V^T rows = channels)
        bfvec16 bv[4];
#pragma unroll
        for (int c = 0; c < 4; ++c)
            bv[c] = load_b_frag(vptr + (c * 16) * NSEQ + kb, NSEQ);
#pragma unroll
        for (int rt = 0; rt < 2; ++rt) {
            bfvec16 pa = load_a_frag(myP + rt * 16 * 32, 32);
#pragma unroll
            for (int c = 0; c < 4; ++c)
                accO[rt][c] = wmma_bf16(pa, bv[c], accO[rt][c]);
        }
        __syncthreads();
    }

    // normalize + store as [B, N, D] bf16
    int b = bh / NH;
    int h = bh - b * NH;
#pragma unroll
    for (int rt = 0; rt < 2; ++rt)
#pragma unroll
        for (int c = 0; c < 4; ++c)
#pragma unroll
            for (int r = 0; r < 8; ++r) {
                int n   = q0 + rt * 16 + rgrp + r;
                float v = accO[rt][c][r] / lrun[rt][r];
                Ob[(b * NSEQ + n) * DIM + h * 64 + c * 16 + cn] = f2bf(v);
            }
}

// ---------------- output projection GEMM (64x64 tile per wave) ----------------
__global__ void __launch_bounds__(256) gemm_proj_kernel(const ushort* __restrict__ abf,
                                                        const ushort* __restrict__ wT,
                                                        const float*  __restrict__ bias,
                                                        float* __restrict__ out) {
    constexpr int TILES_N = DIM / 64;  // 12
    int wave = (blockIdx.x * blockDim.x + threadIdx.x) >> 5;
    int lane = threadIdx.x & 31;
    int tm = wave / TILES_N, tn = wave - tm * TILES_N;
    int m0 = tm * 64, n0 = tn * 64;

    v8f acc[4][4] = {};
    for (int kk = 0; kk < DIM; kk += 32) {
        bfvec16 a[4], b[4];
#pragma unroll
        for (int i = 0; i < 4; ++i)
            a[i] = load_a_frag(abf + (m0 + 16 * i) * DIM + kk, DIM);
#pragma unroll
        for (int j = 0; j < 4; ++j)
            b[j] = load_b_frag(wT + (n0 + 16 * j) * DIM + kk, DIM);
#pragma unroll
        for (int i = 0; i < 4; ++i)
#pragma unroll
            for (int j = 0; j < 4; ++j)
                acc[i][j] = wmma_bf16(a[i], b[j], acc[i][j]);
    }

    int cn   = lane & 15;
    int rgrp = (lane >> 4) * 8;
#pragma unroll
    for (int i = 0; i < 4; ++i)
#pragma unroll
        for (int j = 0; j < 4; ++j) {
            int c = n0 + j * 16 + cn;
            float bv = bias[c];
#pragma unroll
            for (int r = 0; r < 8; ++r) {
                int row = m0 + i * 16 + rgrp + r;
                out[row * DIM + c] = acc[i][j][r] + bv;
            }
        }
}

// ---------------- launch ----------------
extern "C" void kernel_launch(void* const* d_in, const int* in_sizes, int n_in,
                              void* d_out, int out_size, void* d_ws, size_t ws_size,
                              hipStream_t stream) {
    (void)in_sizes; (void)n_in; (void)out_size; (void)ws_size;

    const float* x      = (const float*)d_in[0];
    const float* w_qkv  = (const float*)d_in[1];
    const float* b_qkv  = (const float*)d_in[2];
    const float* w_proj = (const float*)d_in[3];
    const float* b_proj = (const float*)d_in[4];
    float* out = (float*)d_out;

    // workspace carve-up (bf16 elements)
    ushort* p      = (ushort*)d_ws;
    ushort* xbf    = p;  p += (size_t)ROWS * DIM;
    ushort* wqkvT  = p;  p += (size_t)QKVN * DIM;
    ushort* wpT    = p;  p += (size_t)DIM * DIM;
    ushort* Qb     = p;  p += (size_t)BH * NSEQ * HD;
    ushort* Kb     = p;  p += (size_t)BH * NSEQ * HD;
    ushort* Vt     = p;  p += (size_t)BH * HD * NSEQ;
    ushort* Ab     = p;  p += (size_t)ROWS * DIM;

    // 1) fp32 -> bf16 conversions (weights transposed into N x K B-operand layout)
    cvt_bf16_kernel<<<(ROWS * DIM + 255) / 256, 256, 0, stream>>>(x, xbf, ROWS * DIM);
    cvt_transpose_kernel<<<(DIM * QKVN + 255) / 256, 256, 0, stream>>>(w_qkv, wqkvT, DIM, QKVN);
    cvt_transpose_kernel<<<(DIM * DIM + 255) / 256, 256, 0, stream>>>(w_proj, wpT, DIM, DIM);

    // 2) QKV projection: 64 x 36 tiles of 64x64, 8 waves/block
    gemm_qkv_kernel<<<(ROWS / 64) * (QKVN / 64) / 8, 256, 0, stream>>>(
        xbf, wqkvT, b_qkv, Qb, Kb, Vt);

    // 3) flash attention: BH * (NSEQ/256) blocks
    attn_kernel<<<BH * (NSEQ / 256), 256, 0, stream>>>(Qb, Kb, Vt, Ab);

    // 4) output projection -> f32 out (+bias)
    gemm_proj_kernel<<<(ROWS / 64) * (DIM / 64) / 8, 256, 0, stream>>>(
        Ab, wpT, b_proj, out);
}
